// MoEEncoder_78365973283408
// MI455X (gfx1250) — compile-verified, matrix operand order checked by reading, not run on hardware
//
#include <hip/hip_runtime.h>
#include <hip/hip_bf16.h>
#include <math.h>

// ---------------------------------------------------------------------------
// MoE transformer encoder forward for MI455X (gfx1250, wave32, WMMA).
// f16 multiply / f32 accumulate via v_wmma_f32_16x16x32_f16.
// GEMM: 128x64 block tile, 8 waves, 32x32 wave tile (4 WMMA/K-step),
// double-buffered LDS staging with vectorized b128 loads / b64 stores.
// Requirements (hold for every call below): M % 128 == 0, K % 32 == 0, N % 4 == 0.
// ---------------------------------------------------------------------------

typedef _Float16 v4h  __attribute__((ext_vector_type(4)));
typedef _Float16 v8h  __attribute__((ext_vector_type(8)));
typedef _Float16 v16h __attribute__((ext_vector_type(16)));
typedef float    v4f  __attribute__((ext_vector_type(4)));
typedef float    v8f  __attribute__((ext_vector_type(8)));

#define LDSP 40   // halfs per LDS row (80B pitch -> conflict-free b128 reads)

// EPI: 0 = alpha only, 1 = alpha + bias, 2 = alpha + bias + exact GELU
template<int EPI, bool TRANSB>
__global__ __launch_bounds__(256)
void gemm_wmma_kernel(const float* __restrict__ A, long sAb, long sAh, int lda,
                      const float* __restrict__ B, long sBb, long sBh, int ldb,
                      const float* __restrict__ bias,
                      float* __restrict__ C, long sCb, long sCh, int ldc,
                      int M, int N, int Kd, int nh, float alpha)
{
    __shared__ __align__(16) _Float16 lA[2][128][LDSP];
    __shared__ __align__(16) _Float16 lB[2][64][LDSP];   // stored as Bt[n][k]

    const int zb = blockIdx.z / nh, zh = blockIdx.z % nh;
    const float* Ab = A + (long)zb * sAb + (long)zh * sAh;
    const float* Bb = B + (long)zb * sBb + (long)zh * sBh;
    float*       Cb = C + (long)zb * sCb + (long)zh * sCh;

    const int m0 = blockIdx.y * 128;
    const int n0 = blockIdx.x * 64;
    const int tid  = threadIdx.x;
    const int lane = tid & 31;
    const int wid  = tid >> 5;
    const int wm = (wid >> 1) * 32;    // 4 wave-rows x 32
    const int wn = (wid & 1) * 32;     // 2 wave-cols x 32

    // Per-thread staging coordinates (fixed across K-steps).
    // A tile: 128x32 floats = 1024 float4 groups = 4 per thread.
    int aRow[4], aC4[4];
#pragma unroll
    for (int p = 0; p < 4; ++p) {
        int g = p * 256 + tid;
        aRow[p] = g >> 3;
        aC4[p]  = (g & 7) * 4;
    }
    // B tile: 64x32 floats = 512 float4 groups = 2 per thread.
    int bI0[2], bI1[2];   // TRANSB: (n, c4) ; else: (k, n4)
    bool bOk[2];
#pragma unroll
    for (int p = 0; p < 2; ++p) {
        int g = p * 256 + tid;
        if (TRANSB) { bI0[p] = g >> 3;  bI1[p] = (g & 7) * 4;  bOk[p] = (n0 + bI0[p]) < N; }
        else        { bI0[p] = g >> 4;  bI1[p] = (g & 15) * 4; bOk[p] = (n0 + bI1[p]) < N; }
    }

    v8f acc[2][2];
#pragma unroll
    for (int i = 0; i < 2; ++i)
#pragma unroll
        for (int j = 0; j < 2; ++j) acc[i][j] = (v8f){};

    // Fragment addressing (ISA 16-bit 16x32 A / 32x16 B layouts):
    // element e <-> K = e + (e>=8 ? 8 : 0) + (lane>=16 ? 8 : 0)
    const int kb    = (lane >> 4) * 8;
    const int arow0 = wm + (lane & 15);
    const int brow0 = wn + (lane & 15);

    const int nk = Kd >> 5;

    // ---- prologue: stage K-tile 0 into buffer 0 ----
    {
#pragma unroll
        for (int p = 0; p < 4; ++p) {
            v4f t = *(const v4f*)&Ab[(long)(m0 + aRow[p]) * lda + aC4[p]];
            *(v4h*)&lA[0][aRow[p]][aC4[p]] = __builtin_convertvector(t, v4h);
        }
#pragma unroll
        for (int p = 0; p < 2; ++p) {
            v4f t = (v4f){};
            if (TRANSB) {
                if (bOk[p]) t = *(const v4f*)&Bb[(long)(n0 + bI0[p]) * ldb + bI1[p]];
                *(v4h*)&lB[0][bI0[p]][bI1[p]] = __builtin_convertvector(t, v4h);
            } else {
                if (bOk[p]) t = *(const v4f*)&Bb[(long)bI0[p] * ldb + n0 + bI1[p]];
                v4h hv = __builtin_convertvector(t, v4h);
#pragma unroll
                for (int j = 0; j < 4; ++j) lB[0][bI1[p] + j][bI0[p]] = hv[j];
            }
        }
    }
    __syncthreads();

    for (int kk = 0; kk < nk; ++kk) {
        const int cur = kk & 1, nxt = cur ^ 1;
        const bool more = (kk + 1) < nk;
        const long ko = (long)(kk + 1) << 5;

        // ---- issue next tile's global loads (latency hidden under WMMA) ----
        v4f ta[4], tb[2];
        if (more) {
#pragma unroll
            for (int p = 0; p < 4; ++p)
                ta[p] = *(const v4f*)&Ab[(long)(m0 + aRow[p]) * lda + ko + aC4[p]];
#pragma unroll
            for (int p = 0; p < 2; ++p) {
                tb[p] = (v4f){};
                if (TRANSB) {
                    if (bOk[p]) tb[p] = *(const v4f*)&Bb[(long)(n0 + bI0[p]) * ldb + ko + bI1[p]];
                } else {
                    if (bOk[p]) tb[p] = *(const v4f*)&Bb[(ko + bI0[p]) * ldb + n0 + bI1[p]];
                }
            }
        }

        // ---- fragments + 4 WMMAs from current buffer ----
        union F { v16h v; v8h h[2]; };
        F fa[2], fb[2];
#pragma unroll
        for (int i = 0; i < 2; ++i) {
            const _Float16* pa = &lA[cur][arow0 + i * 16][kb];
            fa[i].h[0] = *(const v8h*)pa;
            fa[i].h[1] = *(const v8h*)(pa + 16);
            const _Float16* pb = &lB[cur][brow0 + i * 16][kb];
            fb[i].h[0] = *(const v8h*)pb;
            fb[i].h[1] = *(const v8h*)(pb + 16);
        }
#pragma unroll
        for (int i = 0; i < 2; ++i)
#pragma unroll
            for (int j = 0; j < 2; ++j)
                acc[i][j] = __builtin_amdgcn_wmma_f32_16x16x32_f16(
                    false, fa[i].v, false, fb[j].v, (short)0, acc[i][j], false, false);

        // ---- commit next tile to the ping-pong buffer ----
        if (more) {
#pragma unroll
            for (int p = 0; p < 4; ++p)
                *(v4h*)&lA[nxt][aRow[p]][aC4[p]] = __builtin_convertvector(ta[p], v4h);
#pragma unroll
            for (int p = 0; p < 2; ++p) {
                if (TRANSB) {
                    *(v4h*)&lB[nxt][bI0[p]][bI1[p]] = __builtin_convertvector(tb[p], v4h);
                } else {
                    v4h hv = __builtin_convertvector(tb[p], v4h);
#pragma unroll
                    for (int j = 0; j < 4; ++j) lB[nxt][bI1[p] + j][bI0[p]] = hv[j];
                }
            }
        }
        __syncthreads();
    }

    // ---- epilogue: C/D layout -> VGPR r holds M = r + (lane>=16)*8, N = lane&15
    const int crow = m0 + wm + ((lane >> 4) * 8);
    const int ccol = n0 + wn + (lane & 15);
#pragma unroll
    for (int j = 0; j < 2; ++j) {
        const int col = ccol + j * 16;
        if (col >= N) continue;
        const float bv = (EPI >= 1) ? bias[col] : 0.0f;
#pragma unroll
        for (int i = 0; i < 2; ++i) {
#pragma unroll
            for (int r = 0; r < 8; ++r) {
                const int row = crow + i * 16 + r;
                float v = acc[i][j][r] * alpha + bv;
                if (EPI == 2) v = 0.5f * v * (1.0f + erff(v * 0.70710678118654752f));
                Cb[(long)row * ldc + col] = v;
            }
        }
    }
}

// ---------------------------------------------------------------------------

__global__ void embed_kernel(const int* __restrict__ src, const float* __restrict__ steps,
                             const float* __restrict__ tok, const float* __restrict__ pos,
                             const float* __restrict__ stepe, float* __restrict__ x,
                             int Cc, int Dd, int total)
{
    int i = blockIdx.x * blockDim.x + threadIdx.x;
    if (i >= total) return;
    int d = i % Dd;
    int t = i / Dd;
    int c = t % Cc;
    int b = t / Cc;
    int tk = src[t];
    x[i] = tok[(long)tk * Dd + d] * 22.62741699796952f   // sqrt(512)
         + pos[(long)c * Dd + d] + stepe[d] * steps[b];
}

__global__ __launch_bounds__(256)
void softmax_mask_kernel(float* __restrict__ scores, const int* __restrict__ mask,
                         int NHc, int Cc)
{
    __shared__ float red[256];
    long row = blockIdx.x;                       // rows = B*NH*C
    int  b   = (int)(row / ((long)NHc * Cc));
    float* p = scores + row * Cc;
    const int* mrow = mask + (long)b * Cc;
    int tid = threadIdx.x;

    float lmax = -3.0e38f;
    for (int c = tid; c < Cc; c += 256) {
        float v = (mrow[c] == 0) ? -1.0e10f : p[c];
        p[c] = v;
        lmax = fmaxf(lmax, v);
    }
    red[tid] = lmax; __syncthreads();
    for (int s = 128; s > 0; s >>= 1) {
        if (tid < s) red[tid] = fmaxf(red[tid], red[tid + s]);
        __syncthreads();
    }
    float m = red[0]; __syncthreads();

    float lsum = 0.0f;
    for (int c = tid; c < Cc; c += 256) {
        float e = __expf(p[c] - m);
        p[c] = e;
        lsum += e;
    }
    red[tid] = lsum; __syncthreads();
    for (int s = 128; s > 0; s >>= 1) {
        if (tid < s) red[tid] += red[tid + s];
        __syncthreads();
    }
    float inv = 1.0f / red[0];
    for (int c = tid; c < Cc; c += 256) p[c] *= inv;
}

// x = LayerNorm(x + o) ; one block per token, D = 512, 256 threads
__global__ __launch_bounds__(256)
void add_layernorm_kernel(float* __restrict__ x, const float* __restrict__ o,
                          const float* __restrict__ g, const float* __restrict__ b,
                          int Dd)
{
    __shared__ float s1[256], s2[256];
    long t = blockIdx.x;
    float* xr = x + t * Dd;
    const float* orr = o + t * Dd;
    int tid = threadIdx.x;

    float e0 = xr[tid] + orr[tid];
    float e1 = xr[tid + 256] + orr[tid + 256];
    s1[tid] = e0 + e1;
    s2[tid] = e0 * e0 + e1 * e1;
    __syncthreads();
    for (int s = 128; s > 0; s >>= 1) {
        if (tid < s) { s1[tid] += s1[tid + s]; s2[tid] += s2[tid + s]; }
        __syncthreads();
    }
    float mean = s1[0] / (float)Dd;
    float var  = s2[0] / (float)Dd - mean * mean;
    float rstd = rsqrtf(var + 1e-5f);
    xr[tid]       = (e0 - mean) * rstd * g[tid]       + b[tid];
    xr[tid + 256] = (e1 - mean) * rstd * g[tid + 256] + b[tid + 256];
}

// top-2 of E=8 logits -> sparse softmax gates (tie-break: lower index first)
__global__ void gates_kernel(const float* __restrict__ logits, float* __restrict__ gates,
                             int T, int Ee)
{
    int t = blockIdx.x * blockDim.x + threadIdx.x;
    if (t >= T) return;
    const float* lr = logits + (long)t * Ee;
    float v1 = -3.0e38f, v2 = -3.0e38f; int i1 = 0, i2 = 0;
    for (int e = 0; e < Ee; ++e) {
        float v = lr[e];
        if (v > v1)      { v2 = v1; i2 = i1; v1 = v; i1 = e; }
        else if (v > v2) { v2 = v; i2 = e; }
    }
    float e2 = __expf(v2 - v1);
    float inv = 1.0f / (1.0f + e2);
    float* gr = gates + (long)t * Ee;
    for (int e = 0; e < Ee; ++e) gr[e] = 0.0f;
    gr[i1] = inv;
    gr[i2] = e2 * inv;
}

__global__ void moe_accum_kernel(const float* __restrict__ ye, const float* __restrict__ gates,
                                 float* __restrict__ moe, int Dd, int Ee, int e,
                                 float beta, int total)
{
    int i = blockIdx.x * blockDim.x + threadIdx.x;
    if (i >= total) return;
    int t = i / Dd;
    moe[i] = beta * moe[i] + gates[(long)t * Ee + e] * ye[i];
}

__global__ void copy_kernel(const float* __restrict__ src, float* __restrict__ dst, int total)
{
    int i = blockIdx.x * blockDim.x + threadIdx.x;
    if (i < total) dst[i] = src[i];
}

// ---------------------------------------------------------------------------

static void gemm(int epi, bool transB,
                 const float* A, long sAb, long sAh, int lda,
                 const float* B, long sBb, long sBh, int ldb,
                 const float* bias,
                 float* C, long sCb, long sCh, int ldc,
                 int M, int N, int Kd, int batches, int nh, float alpha,
                 hipStream_t stream)
{
    dim3 grid((N + 63) / 64, M / 128, batches);
    dim3 blk(256);
    if (transB) {
        gemm_wmma_kernel<0, true><<<grid, blk, 0, stream>>>(
            A, sAb, sAh, lda, B, sBb, sBh, ldb, bias, C, sCb, sCh, ldc, M, N, Kd, nh, alpha);
    } else if (epi == 0) {
        gemm_wmma_kernel<0, false><<<grid, blk, 0, stream>>>(
            A, sAb, sAh, lda, B, sBb, sBh, ldb, bias, C, sCb, sCh, ldc, M, N, Kd, nh, alpha);
    } else if (epi == 1) {
        gemm_wmma_kernel<1, false><<<grid, blk, 0, stream>>>(
            A, sAb, sAh, lda, B, sBb, sBh, ldb, bias, C, sCb, sCh, ldc, M, N, Kd, nh, alpha);
    } else {
        gemm_wmma_kernel<2, false><<<grid, blk, 0, stream>>>(
            A, sAb, sAh, lda, B, sBb, sBh, ldb, bias, C, sCb, sCh, ldc, M, N, Kd, nh, alpha);
    }
}

extern "C" void kernel_launch(void* const* d_in, const int* in_sizes, int n_in,
                              void* d_out, int out_size, void* d_ws, size_t ws_size,
                              hipStream_t stream)
{
    (void)in_sizes; (void)n_in; (void)out_size; (void)ws_size;

    const int Bc = 2, Cc = 512, Dd = 512, NHc = 8, Ee = 8, FFc = 2048, Ll = 2, HDc = 64;
    const int T  = Bc * Cc;
    const long TD = (long)T * Dd;

    const int*   src      = (const int*)  d_in[0];
    const int*   mask     = (const int*)  d_in[1];
    const float* steps    = (const float*)d_in[2];
    const float* tok_emb  = (const float*)d_in[3];
    const float* pos_emb  = (const float*)d_in[4];
    const float* step_emb = (const float*)d_in[5];
    const float* aln_g    = (const float*)d_in[6];
    const float* aln_b    = (const float*)d_in[7];
    const float* fln_g    = (const float*)d_in[8];
    const float* fln_b    = (const float*)d_in[9];
    const float* Wq = (const float*)d_in[10];
    const float* Wk = (const float*)d_in[11];
    const float* Wv = (const float*)d_in[12];
    const float* Wo = (const float*)d_in[13];
    const float* bq = (const float*)d_in[14];
    const float* bk = (const float*)d_in[15];
    const float* bv = (const float*)d_in[16];
    const float* bo = (const float*)d_in[17];
    const float* rW = (const float*)d_in[18];
    const float* rb = (const float*)d_in[19];
    const float* eW1 = (const float*)d_in[20];
    const float* eb1 = (const float*)d_in[21];
    const float* eW2 = (const float*)d_in[22];
    const float* eb2 = (const float*)d_in[23];

    float* ws = (float*)d_ws;
    float* x      = ws;               // TD
    float* q      = x + TD;           // TD
    float* k      = q + TD;           // TD
    float* v      = k + TD;           // TD
    float* attno  = v + TD;           // TD
    float* proj   = attno + TD;       // TD
    float* ye     = proj + TD;        // TD
    float* moe    = ye + TD;          // TD
    float* scores = moe + TD;         // B*NH*C*C = 4,194,304
    float* h      = scores + (long)Bc * NHc * Cc * Cc;  // T*FF
    float* logits = h + (long)T * FFc;                  // T*E
    float* gates  = logits + (long)T * Ee;              // T*E

    // --- embedding ---
    {
        int total = (int)TD;
        embed_kernel<<<(total + 255) / 256, 256, 0, stream>>>(
            src, steps, tok_emb, pos_emb, step_emb, x, Cc, Dd, total);
    }

    for (int l = 0; l < Ll; ++l) {
        const float* Wql = Wq + (long)l * Dd * Dd;
        const float* Wkl = Wk + (long)l * Dd * Dd;
        const float* Wvl = Wv + (long)l * Dd * Dd;
        const float* Wol = Wo + (long)l * Dd * Dd;

        // Q, K, V projections
        gemm(1, false, x, 0, 0, Dd, Wql, 0, 0, Dd, bq + (long)l * Dd,
             q, 0, 0, Dd, T, Dd, Dd, 1, 1, 1.0f, stream);
        gemm(1, false, x, 0, 0, Dd, Wkl, 0, 0, Dd, bk + (long)l * Dd,
             k, 0, 0, Dd, T, Dd, Dd, 1, 1, 1.0f, stream);
        gemm(1, false, x, 0, 0, Dd, Wvl, 0, 0, Dd, bv + (long)l * Dd,
             v, 0, 0, Dd, T, Dd, Dd, 1, 1, 1.0f, stream);

        // scores = Q @ K^T / sqrt(HD), batched over (b, h)
        gemm(0, true,
             q, (long)Cc * Dd, HDc, Dd,
             k, (long)Cc * Dd, HDc, Dd, nullptr,
             scores, (long)NHc * Cc * Cc, (long)Cc * Cc, Cc,
             Cc, Cc, HDc, Bc * NHc, NHc, 0.125f, stream);

        softmax_mask_kernel<<<Bc * NHc * Cc, 256, 0, stream>>>(scores, mask, NHc, Cc);

        // attn @ V, batched over (b, h)
        gemm(0, false,
             scores, (long)NHc * Cc * Cc, (long)Cc * Cc, Cc,
             v, (long)Cc * Dd, HDc, Dd, nullptr,
             attno, (long)Cc * Dd, HDc, Dd,
             Cc, HDc, Cc, Bc * NHc, NHc, 1.0f, stream);

        // output projection
        gemm(1, false, attno, 0, 0, Dd, Wol, 0, 0, Dd, bo + (long)l * Dd,
             proj, 0, 0, Dd, T, Dd, Dd, 1, 1, 1.0f, stream);

        add_layernorm_kernel<<<T, 256, 0, stream>>>(x, proj,
            aln_g + (long)l * Dd, aln_b + (long)l * Dd, Dd);

        // --- MoE ---
        gemm(1, false, x, 0, 0, Dd, rW + (long)l * Dd * Ee, 0, 0, Ee,
             rb + (long)l * Ee, logits, 0, 0, Ee, T, Ee, Dd, 1, 1, 1.0f, stream);
        gates_kernel<<<(T + 255) / 256, 256, 0, stream>>>(logits, gates, T, Ee);

        for (int e = 0; e < Ee; ++e) {
            const float* W1 = eW1 + ((long)l * Ee + e) * Dd * FFc;
            const float* b1 = eb1 + ((long)l * Ee + e) * FFc;
            const float* W2 = eW2 + ((long)l * Ee + e) * FFc * Dd;
            const float* b2 = eb2 + ((long)l * Ee + e) * Dd;
            // h = GELU(x @ W1 + b1)
            gemm(2, false, x, 0, 0, Dd, W1, 0, 0, FFc, b1,
                 h, 0, 0, FFc, T, FFc, Dd, 1, 1, 1.0f, stream);
            // ye = h @ W2 + b2
            gemm(1, false, h, 0, 0, FFc, W2, 0, 0, Dd, b2,
                 ye, 0, 0, Dd, T, Dd, FFc, 1, 1, 1.0f, stream);
            // moe (+)= gate[:, e] * ye
            moe_accum_kernel<<<((int)TD + 255) / 256, 256, 0, stream>>>(
                ye, gates, moe, Dd, Ee, e, (e == 0) ? 0.0f : 1.0f, (int)TD);
        }

        add_layernorm_kernel<<<T, 256, 0, stream>>>(x, moe,
            fln_g + (long)l * Dd, fln_b + (long)l * Dd, Dd);
    }

    copy_kernel<<<((int)TD + 255) / 256, 256, 0, stream>>>(x, (float*)d_out, (int)TD);
}
